// UltimateFusionV5_48979807043622
// MI455X (gfx1250) — compile-verified, matrix operand order checked by reading, not run on hardware
//
#include <hip/hip_runtime.h>
#include <math.h>

// ---------------- problem constants (from reference) ----------------
#define BB   8
#define SS   2048
#define DD   512
#define NB   8
#define KSEL 2
#define BS_ROWS (BB * SS)            // 16384 rows
#define OUT_ROWSTRIDE (2 * SS * DD)  // per-batch stride in d_out (B, 2S, D)

// Use gfx1250 async global->LDS copies in the GEMM staging path.
#define USE_ASYNC 1

// ---------------- CDNA5 vector types ----------------
typedef __attribute__((ext_vector_type(16))) __bf16 v16bf;
typedef __attribute__((ext_vector_type(8)))  float  v8f;
typedef __attribute__((ext_vector_type(4)))  unsigned int u32x4;

union FragBF { v16bf v; u32x4 q[2]; };

__device__ inline v8f zero8() { v8f z = {0.f,0.f,0.f,0.f,0.f,0.f,0.f,0.f}; return z; }

// round-to-nearest-even f32 -> bf16
__device__ inline __bf16 f2bf(float f) {
    union { float f; unsigned int u; } in; in.f = f;
    unsigned int u = in.u;
    unsigned int r = (u + 0x7FFFu + ((u >> 16) & 1u)) >> 16;
    union { unsigned short s; __bf16 b; } cv; cv.s = (unsigned short)r;
    return cv.b;
}

__device__ inline float gelu_exact(float x) {
    return 0.5f * x * (1.0f + erff(x * 0.70710678118654752f));
}

// ---------------- column mean over S: (B,S,D) -> (B,D) ----------------
__global__ __launch_bounds__(256) void colmean_kernel(const float* __restrict__ in,
                                                      float* __restrict__ outm) {
    int gid = blockIdx.x * blockDim.x + threadIdx.x;  // B*D threads
    int b = gid / DD, d = gid % DD;
    const float* p = in + (size_t)b * SS * DD + d;
    float s = 0.f;
    #pragma unroll 4
    for (int i = 0; i < SS; ++i) s += p[(size_t)i * DD];
    outm[gid] = s * (1.0f / SS);
}

// ---------------- selector: sigmoid(mean[0] @ W + b), top-K of NB ----------------
__global__ void select_topk_kernel(const float* __restrict__ meanAll,  // row 0 used
                                   const float* __restrict__ selW,    // D x NB
                                   const float* __restrict__ selB,    // NB
                                   int* __restrict__ topOut) {        // KSEL ints
    __shared__ float sc[NB];
    int j = threadIdx.x;
    if (j < NB) {
        float s = selB[j];
        for (int d = 0; d < DD; ++d) s += meanAll[d] * selW[d * NB + j];
        sc[j] = 1.0f / (1.0f + expf(-s));
    }
    __syncthreads();
    if (threadIdx.x == 0) {
        bool used[NB];
        for (int q = 0; q < NB; ++q) used[q] = false;
        for (int k = 0; k < KSEL; ++k) {
            int bi = 0; float bv = -1e30f;
            for (int q = 0; q < NB; ++q)
                if (!used[q] && sc[q] > bv) { bv = sc[q]; bi = q; }  // strict >: JAX tie rule
            used[bi] = true;
            topOut[k] = bi;
        }
    }
}

// ---------------- init: out[:, :S, :] = embodied, out[:, S:, :] = disembodied ----------------
__global__ __launch_bounds__(256) void init_copy_kernel(const float* __restrict__ emb,
                                                        const float* __restrict__ dis,
                                                        float* __restrict__ out) {
    size_t e = (size_t)blockIdx.x * blockDim.x + threadIdx.x;  // B*2S*D threads
    size_t b = e / OUT_ROWSTRIDE;
    size_t rem = e % OUT_ROWSTRIDE;
    size_t s2 = rem / DD, d = rem % DD;
    float v = (s2 < SS) ? emb[b * SS * DD + s2 * DD + d]
                        : dis[b * SS * DD + (s2 - SS) * DD + d];
    out[e] = v;
}

// ---------------- weight convert+transpose: (NB,K,N) f32 -> (NB,N,K) bf16 ----------------
__global__ __launch_bounds__(256) void wt_convert_kernel(const float* __restrict__ W,
                                                         __bf16* __restrict__ Wt,
                                                         int Kd, int Nd) {
    __shared__ float tile[32][33];
    const int i  = blockIdx.z;
    const int k0 = blockIdx.x * 32, n0 = blockIdx.y * 32;
    const int tx = threadIdx.x & 31, ty = threadIdx.x >> 5;  // 32 x 8
    const float* Wi  = W  + (size_t)i * Kd * Nd;
    __bf16*      Wti = Wt + (size_t)i * Kd * Nd;
    #pragma unroll
    for (int r = ty; r < 32; r += 8)
        tile[r][tx] = Wi[(size_t)(k0 + r) * Nd + n0 + tx];
    __syncthreads();
    #pragma unroll
    for (int r = ty; r < 32; r += 8)
        Wti[(size_t)(n0 + r) * Kd + k0 + tx] = f2bf(tile[tx][r]);
}

// ---------------- LayerNorm over D: block per row, writes bf16 ----------------
__global__ __launch_bounds__(256) void ln_kernel(const float* __restrict__ outBase, int sOff,
                                                 const float* __restrict__ gAll,
                                                 const float* __restrict__ betaAll,
                                                 const int* __restrict__ topIdx, int kStep,
                                                 __bf16* __restrict__ xbuf) {
    int row = blockIdx.x;                 // 0 .. B*S-1
    int b = row / SS, s = row % SS;
    const float* src = outBase + (size_t)b * OUT_ROWSTRIDE + (size_t)(s + sOff) * DD;
    int idx = topIdx[kStep];
    const float* g  = gAll + (size_t)idx * DD;
    const float* bt = betaAll + (size_t)idx * DD;

    int t = threadIdx.x;
    float v0 = src[t], v1 = src[t + 256];
    __shared__ float rs[256], rq[256];
    rs[t] = v0 + v1;
    rq[t] = v0 * v0 + v1 * v1;
    __syncthreads();
    for (int off = 128; off > 0; off >>= 1) {
        if (t < off) { rs[t] += rs[t + off]; rq[t] += rq[t + off]; }
        __syncthreads();
    }
    float m   = rs[0] * (1.0f / DD);
    float var = rq[0] * (1.0f / DD) - m * m;
    float inv = rsqrtf(var + 1e-5f);
    __bf16* dstp = xbuf + (size_t)row * DD;
    dstp[t]       = f2bf((v0 - m) * inv * g[t]       + bt[t]);
    dstp[t + 256] = f2bf((v1 - m) * inv * g[t + 256] + bt[t + 256]);
}

// ---------------- async global->LDS helpers (CDNA5) ----------------
__device__ inline void async_copy_b128(const __bf16* gsrc, const __bf16* ldst) {
#if USE_ASYNC
    unsigned lds  = (unsigned)(size_t)ldst;                 // generic LDS ptr low 32b = LDS offset
    unsigned long long ga = (unsigned long long)(size_t)gsrc;
    asm volatile("global_load_async_to_lds_b128 %0, %1, off"
                 :: "v"(lds), "v"(ga) : "memory");
#else
    *(u32x4*)ldst = *(const u32x4*)gsrc;
#endif
}

__device__ inline void async_wait_all() {
#if USE_ASYNC
    asm volatile("s_wait_asynccnt 0x0" ::: "memory");
#endif
}

// ---------------- bf16 WMMA GEMM with fused epilogues ----------------
// C[16384, N] = A[16384, Kdim](bf16) @ Wt[idx][N, K](bf16, pre-transposed)
// EPI 0: dst = bf16(tanh(acc+bias))
// EPI 1: hn = (acc+bias)*(1+0.1*t);  out += 0.3*hn
// EPI 2: hp = acc+bias+0.2*delayed; hp*=(1+0.05*t); out = out+0.5*hp
// EPI 3: dst = bf16(gelu(acc+bias))
// EPI 4: out += 0.5*(acc+bias)
template <int EPI>
__global__ __launch_bounds__(256) void gemm_bf16_wmma(
        const __bf16* __restrict__ A, int Kdim,
        const __bf16* __restrict__ WtAll, const float* __restrict__ biasAll, int N,
        const int* __restrict__ topIdx, int kStep,
        __bf16* __restrict__ dstBuf,
        float* __restrict__ outBase, int sOff,
        const float* __restrict__ torsion,
        const float* __restrict__ delayed) {
    constexpr int BM = 64, BN = 128, BK = 32;
    __shared__ __align__(16) __bf16 As[2][BM][BK];   // rows x K (4KB/buf)
    __shared__ __align__(16) __bf16 Bs[2][BN][BK];   // cols(N) x K (8KB/buf)

    const int idx = topIdx[kStep];
    const __bf16* __restrict__ Wblk = WtAll + (size_t)idx * Kdim * N;  // [N][K]
    const float*  __restrict__ bias = biasAll + (size_t)idx * N;

    const int rowBase = blockIdx.y * BM;
    const int nBase   = blockIdx.x * BN;

    const int tid  = threadIdx.x;
    const int lane = tid & 31;
    const int wid  = tid >> 5;          // 0..7
    const int waveM = (wid & 1) * 32;   // 2 wave-rows
    const int waveN = (wid >> 1) * 32;  // 4 wave-cols

    // staging maps (all 16B chunks, data already bf16):
    // A tile 64x32 = 256 chunks -> 1 per thread
    const int arow = tid >> 2;                // 0..63
    const int akp  = (tid & 3) * 8;           // elem 0,8,16,24
    // B tile 128x32 = 512 chunks -> 2 per thread (elem pairs {0,16} / {8,24})
    const int bn   = tid >> 1;                // 0..127
    const int bkp  = (tid & 1) * 8;           // elem 0 or 8

    const int nkt = Kdim / BK;

    auto stage = [&](int kt, int buf) {
        const int k0 = kt * BK;
        async_copy_b128(A + (size_t)(rowBase + arow) * Kdim + k0 + akp, &As[buf][arow][akp]);
        const __bf16* wrow = Wblk + (size_t)(nBase + bn) * Kdim + k0;
        async_copy_b128(wrow + bkp,      &Bs[buf][bn][bkp]);
        async_copy_b128(wrow + bkp + 16, &Bs[buf][bn][bkp + 16]);
    };

    v8f acc[4];
    #pragma unroll
    for (int i = 0; i < 4; ++i) acc[i] = zero8();

    const int fr = lane & 15;                 // row (A) / col (B) within 16
    const int kb = (lane < 16) ? 0 : 8;       // K sub-base per ISA 16-bit layouts

    stage(0, 0);
    for (int kt = 0; kt < nkt; ++kt) {
        async_wait_all();
        __syncthreads();
        if (kt + 1 < nkt) stage(kt + 1, (kt + 1) & 1);
        const int cur = kt & 1;

        FragBF fa[2], fb[2];
        #pragma unroll
        for (int mi = 0; mi < 2; ++mi) {
            const __bf16* p = &As[cur][waveM + mi * 16 + fr][0];
            fa[mi].q[0] = *reinterpret_cast<const u32x4*>(p + kb);
            fa[mi].q[1] = *reinterpret_cast<const u32x4*>(p + kb + 16);
        }
        #pragma unroll
        for (int ni = 0; ni < 2; ++ni) {
            const __bf16* p = &Bs[cur][waveN + ni * 16 + fr][0];
            fb[ni].q[0] = *reinterpret_cast<const u32x4*>(p + kb);
            fb[ni].q[1] = *reinterpret_cast<const u32x4*>(p + kb + 16);
        }
        #pragma unroll
        for (int mi = 0; mi < 2; ++mi)
            #pragma unroll
            for (int ni = 0; ni < 2; ++ni)
                acc[mi * 2 + ni] = __builtin_amdgcn_wmma_f32_16x16x32_bf16(
                    false, fa[mi].v, false, fb[ni].v, (short)0, acc[mi * 2 + ni],
                    false, false);
    }

    // ---------------- epilogue (f32 C layout: n = lane&15, m = v + 8*(lane>>4)) ----
    const int half = lane >> 4;
    const int nlo  = lane & 15;
    #pragma unroll
    for (int mi = 0; mi < 2; ++mi) {
        #pragma unroll
        for (int ni = 0; ni < 2; ++ni) {
            v8f a = acc[mi * 2 + ni];
            const int gc = nBase + waveN + ni * 16 + nlo;
            const float bia = bias[gc];
            #pragma unroll
            for (int v = 0; v < 8; ++v) {
                const int gr = rowBase + waveM + mi * 16 + v + 8 * half;
                float val = a[v] + bia;
                if (EPI == 0) {
                    dstBuf[(size_t)gr * N + gc] = f2bf(tanhf(val));
                } else if (EPI == 3) {
                    dstBuf[(size_t)gr * N + gc] = f2bf(gelu_exact(val));
                } else {
                    const int b = gr / SS, s = gr % SS;
                    float* o = outBase + (size_t)b * OUT_ROWSTRIDE +
                               (size_t)(s + sOff) * DD + gc;
                    if (EPI == 1) {
                        val *= (1.0f + 0.1f * torsion[b * DD + gc]);
                        *o = *o + 0.3f * val;
                    } else if (EPI == 2) {
                        val += 0.2f * delayed[b * DD + gc];
                        val *= (1.0f + 0.05f * torsion[b * DD + gc]);
                        *o = *o + 0.5f * val;
                    } else {  // EPI == 4
                        *o = *o + 0.5f * val;
                    }
                }
            }
        }
    }
}

// ---------------- host-side orchestration ----------------
extern "C" void kernel_launch(void* const* d_in, const int* in_sizes, int n_in,
                              void* d_out, int out_size, void* d_ws, size_t ws_size,
                              hipStream_t stream) {
    const float* emb    = (const float*)d_in[0];
    const float* dis    = (const float*)d_in[1];
    const float* tors   = (const float*)d_in[2];
    const float* selEW  = (const float*)d_in[3];
    const float* selEb  = (const float*)d_in[4];
    const float* selDW  = (const float*)d_in[5];
    const float* selDb  = (const float*)d_in[6];
    const float* e_g    = (const float*)d_in[7];
    const float* e_beta = (const float*)d_in[8];
    const float* e_W1   = (const float*)d_in[9];
    const float* e_b1   = (const float*)d_in[10];
    const float* e_W2   = (const float*)d_in[11];
    const float* e_b2   = (const float*)d_in[12];
    const float* d_g1   = (const float*)d_in[13];
    const float* d_be1  = (const float*)d_in[14];
    const float* d_Wa   = (const float*)d_in[15];
    const float* d_ba   = (const float*)d_in[16];
    const float* d_g2   = (const float*)d_in[17];
    const float* d_be2  = (const float*)d_in[18];
    const float* d_Wf1  = (const float*)d_in[19];
    const float* d_bf1  = (const float*)d_in[20];
    const float* d_Wf2  = (const float*)d_in[21];
    const float* d_bf2  = (const float*)d_in[22];
    float* out = (float*)d_out;

    // workspace carve-up
    float* meanE   = (float*)d_ws;             // B*D
    float* delayed = meanE + BB * DD;          // B*D
    int*   topE    = (int*)(delayed + BB * DD);
    int*   topD    = topE + 8;
    __bf16* xbuf   = (__bf16*)((char*)d_ws + 65536);      // 16384 x 512  bf16
    __bf16* ybuf   = xbuf  + (size_t)BS_ROWS * DD;        // 16384 x 1024 bf16
    __bf16* eW1t   = ybuf  + (size_t)BS_ROWS * 2 * DD;    // NB x 512 x 512
    __bf16* eW2t   = eW1t  + (size_t)NB * DD * DD;
    __bf16* dWat   = eW2t  + (size_t)NB * DD * DD;
    __bf16* dWf1t  = dWat  + (size_t)NB * DD * DD;        // NB x 1024(N) x 512(K)
    __bf16* dWf2t  = dWf1t + (size_t)NB * DD * 2 * DD;    // NB x 512(N) x 1024(K)

    colmean_kernel<<<(BB * DD) / 256, 256, 0, stream>>>(emb, meanE);
    colmean_kernel<<<(BB * DD) / 256, 256, 0, stream>>>(dis, delayed);
    select_topk_kernel<<<1, 32, 0, stream>>>(meanE,   selEW, selEb, topE);
    select_topk_kernel<<<1, 32, 0, stream>>>(delayed, selDW, selDb, topD);
    init_copy_kernel<<<(BB * 2 * SS * DD) / 256, 256, 0, stream>>>(emb, dis, out);

    // convert+transpose all block weights to bf16 [N][K]
    wt_convert_kernel<<<dim3(DD/32, DD/32, NB),       256, 0, stream>>>(e_W1,  eW1t,  DD, DD);
    wt_convert_kernel<<<dim3(DD/32, DD/32, NB),       256, 0, stream>>>(e_W2,  eW2t,  DD, DD);
    wt_convert_kernel<<<dim3(DD/32, DD/32, NB),       256, 0, stream>>>(d_Wa,  dWat,  DD, DD);
    wt_convert_kernel<<<dim3(DD/32, (2*DD)/32, NB),   256, 0, stream>>>(d_Wf1, dWf1t, DD, 2*DD);
    wt_convert_kernel<<<dim3((2*DD)/32, DD/32, NB),   256, 0, stream>>>(d_Wf2, dWf2t, 2*DD, DD);

    const dim3 blk(256);
    const dim3 g512(512 / 128, BS_ROWS / 64);   // N=512
    const dim3 g1024(1024 / 128, BS_ROWS / 64); // N=1024

    // embodied scan: h += 0.3 * [(tanh(LN(h)@W1+b1)@W2+b2) * (1+0.1 t)]
    for (int k = 0; k < KSEL; ++k) {
        ln_kernel<<<BS_ROWS, blk, 0, stream>>>(out, 0, e_g, e_beta, topE, k, xbuf);
        gemm_bf16_wmma<0><<<g512, blk, 0, stream>>>(xbuf, DD, eW1t, e_b1, DD, topE, k,
                                                    ybuf, nullptr, 0, nullptr, nullptr);
        gemm_bf16_wmma<1><<<g512, blk, 0, stream>>>(ybuf, DD, eW2t, e_b2, DD, topE, k,
                                                    nullptr, out, 0, tors, nullptr);
    }

    // disembodied scan
    for (int k = 0; k < KSEL; ++k) {
        ln_kernel<<<BS_ROWS, blk, 0, stream>>>(out, SS, d_g1, d_be1, topD, k, xbuf);
        gemm_bf16_wmma<2><<<g512, blk, 0, stream>>>(xbuf, DD, dWat, d_ba, DD, topD, k,
                                                    nullptr, out, SS, tors, delayed);
        ln_kernel<<<BS_ROWS, blk, 0, stream>>>(out, SS, d_g2, d_be2, topD, k, xbuf);
        gemm_bf16_wmma<3><<<g1024, blk, 0, stream>>>(xbuf, DD, dWf1t, d_bf1, 2 * DD, topD, k,
                                                     ybuf, nullptr, 0, nullptr, nullptr);
        gemm_bf16_wmma<4><<<g512, blk, 0, stream>>>(ybuf, 2 * DD, dWf2t, d_bf2, DD, topD, k,
                                                    nullptr, out, SS, nullptr, nullptr);
    }
}
